// GINLayer_36283883717329
// MI455X (gfx1250) — matrix operand order repaired and checked by reading, not compiled, not run on hardware
//
#include <hip/hip_runtime.h>
#include <hip/hip_bf16.h>

#define N_NODES 50000
#define N_EDGES 800000
#define D_IN 128
#define D_HID 256
#define D_OUT 128

typedef __bf16 bf16_t;
typedef float  v4f  __attribute__((ext_vector_type(4)));
typedef float  v8f  __attribute__((ext_vector_type(8)));
typedef bf16_t v8bf __attribute__((ext_vector_type(8)));
typedef bf16_t v16bf __attribute__((ext_vector_type(16)));

// ---------------------------------------------------------------------------
// Kernel 0: h0 = (1 + eps) * x     (fp32, workspace)
// ---------------------------------------------------------------------------
__global__ void k_init_h0(const float* __restrict__ x, const float* __restrict__ eps,
                          float* __restrict__ h0) {
  int i = blockIdx.x * blockDim.x + threadIdx.x;   // over float4 elements
  float s = 1.0f + eps[0];
  if (i < N_NODES * D_IN / 4) {
    v4f v = ((const v4f*)x)[i];
    ((v4f*)h0)[i] = v * s;
  }
}

// ---------------------------------------------------------------------------
// Kernel 1: pack row-major f32 weight [K][N] into bf16 WMMA B fragments.
// B layout (CDNA5 ISA): lanes 0-15 hold K=0..7,16..23 (j<8: K=j, j>=8: K=j+8),
// lanes 16-31 hold K=8..15,24..31; column n = nt*16 + (lane & 15).
// Storage lane-major: out[frag*512 + lane*16 + j] -> 32 B contiguous per lane.
// ---------------------------------------------------------------------------
__global__ void k_pack_w(const float* __restrict__ w, int K, int N,
                         bf16_t* __restrict__ out) {
  int f = blockIdx.x;            // one wave per fragment
  int lane = threadIdx.x;
  int ntiles = N >> 4;
  int nt = f % ntiles, kt = f / ntiles;
  int n = (nt << 4) + (lane & 15);
  int kbase = (kt << 5) + ((lane >> 4) << 3);      // +8 for upper lane half
  bf16_t* dst = out + (size_t)f * 512 + lane * 16;
#pragma unroll
  for (int j = 0; j < 16; ++j)
    dst[j] = (bf16_t)w[(size_t)(kbase + j + (j & 8)) * N + n];
}

// ---------------------------------------------------------------------------
// Kernel 2: edge aggregation. edge_dst sorted -> run-accumulate in registers,
// flush with global_atomic_add_f32 at segment boundaries. x rows L2-resident.
// Next-edge metadata prefetched to overlap with the x gather.
// ---------------------------------------------------------------------------
#define AGG_BLOCKS 512
#define AGG_THREADS 256
#define AGG_WAVES (AGG_BLOCKS * AGG_THREADS / 32)

__global__ void k_edge_agg(const float* __restrict__ x, const int* __restrict__ esrc,
                           const int* __restrict__ edst, const float* __restrict__ ev,
                           float* __restrict__ h0) {
  int wave = (blockIdx.x * blockDim.x + threadIdx.x) >> 5;
  int lane = threadIdx.x & 31;
  const int chunk = (N_EDGES + AGG_WAVES - 1) / AGG_WAVES;
  int e0 = wave * chunk;
  int e1 = min(e0 + chunk, N_EDGES);
  if (e0 >= e1) return;
  const v4f vzero = {0.f, 0.f, 0.f, 0.f};
  v4f acc = vzero;
  int d = edst[e0];
  int s = esrc[e0];
  float v = ev[e0];
  int cur = d;
  for (int e = e0; e < e1; ++e) {
    int nd = 0, ns = 0;
    float nv = 0.f;
    if (e + 1 < e1) {            // prefetch next edge metadata
      nd = edst[e + 1];
      ns = esrc[e + 1];
      nv = ev[e + 1];
    }
    if (d != cur) {              // wave-uniform: no divergence
      float* p = h0 + (size_t)cur * D_IN + lane * 4;
      atomicAdd(p + 0, acc.x);
      atomicAdd(p + 1, acc.y);
      atomicAdd(p + 2, acc.z);
      atomicAdd(p + 3, acc.w);
      acc = vzero;
      cur = d;
    }
    v4f xv = ((const v4f*)(x + (size_t)s * D_IN))[lane];
    acc += v * xv;
    d = nd; s = ns; v = nv;
  }
  float* p = h0 + (size_t)cur * D_IN + lane * 4;
  atomicAdd(p + 0, acc.x);
  atomicAdd(p + 1, acc.y);
  atomicAdd(p + 2, acc.z);
  atomicAdd(p + 3, acc.w);
}

// ---------------------------------------------------------------------------
// Kernel 3: fused MLP, 32 rows (2 x 16-row sub-blocks) per wave; each B
// fragment feeds two WMMAs. Per K-tile, all 8 B fragments are loaded into a
// register array FIRST (one clause of 16 global_load_b128), then 16 WMMAs
// consume them -> load latency hidden behind 16 WMMAs instead of 2.
// Live VGPRs ~210 < 256 (no s_set_vgpr_msb juggling).
// A layout (CDNA5 ISA): lane<16 holds M=lane, j=0..7 -> K=+0..7,
// j=8..15 -> K=+16..23; lanes>=16 K-base +8. D layout: VGPR r, lanes 0-15 ->
// (M=r, N=lane); lanes 16-31 -> (M=r+8, N=lane-16).
// ---------------------------------------------------------------------------
#define MLP_WPB 2
#define ROW_TASKS ((N_NODES + 31) / 32)   // 1563; last task has only 16 rows

__device__ __forceinline__ v16bf load_afrag_f32(const float* p) {
  v4f a0 = *(const v4f*)(p);
  v4f a1 = *(const v4f*)(p + 4);
  v4f a2 = *(const v4f*)(p + 16);
  v4f a3 = *(const v4f*)(p + 20);
  v16bf r;
#pragma unroll
  for (int j = 0; j < 4; ++j) {
    r[j]      = (bf16_t)a0[j];
    r[j + 4]  = (bf16_t)a1[j];
    r[j + 8]  = (bf16_t)a2[j];
    r[j + 12] = (bf16_t)a3[j];
  }
  return r;
}

__device__ __forceinline__ v16bf load_afrag_lds(const bf16_t* p) {
  v8bf lo = *(const v8bf*)(p);        // K base+0..7
  v8bf hi = *(const v8bf*)(p + 16);   // K base+16..23
  return __builtin_shufflevector(lo, hi, 0, 1, 2, 3, 4, 5, 6, 7,
                                 8, 9, 10, 11, 12, 13, 14, 15);
}

__global__ void __launch_bounds__(32 * MLP_WPB)
k_mlp(const float* __restrict__ h0, const bf16_t* __restrict__ w1p,
      const float* __restrict__ b1, const bf16_t* __restrict__ w2p,
      const float* __restrict__ b2, float* __restrict__ out) {
  __shared__ __align__(16) bf16_t smem[MLP_WPB][32 * D_HID];   // 2 x 16 KB
  int wid = threadIdx.x >> 5;
  int lane = threadIdx.x & 31;
  int task = blockIdx.x * MLP_WPB + wid;      // wave-uniform
  if (task >= ROW_TASKS) return;              // whole-wave exit, EXEC stays full
  int row0 = task * 32;
  bool full = (row0 + 32 <= N_NODES);         // tail: only first 16 rows valid
  int l16 = lane & 15;
  int khalf = (lane >> 4) << 3;               // 0 or 8 (A k-offset AND D row-offset)

  const v8f zero8 = {0.f, 0.f, 0.f, 0.f, 0.f, 0.f, 0.f, 0.f};
  bf16_t* sc = smem[wid];                     // [32 rows][256 cols] bf16

  const float* arow0 = h0 + (size_t)(row0 + l16) * D_IN;
  const float* arow1 = h0 + (size_t)(row0 + 16 + l16) * D_IN;  // tail: reads
  // land in the packed-weight region of ws (in-bounds, values discarded)

  // ---------------- layer 1: two groups of 8 hidden N-tiles ----------------
#pragma unroll
  for (int g = 0; g < 2; ++g) {
    v8f a0[8], a1[8];
#pragma unroll
    for (int n = 0; n < 8; ++n) { a0[n] = zero8; a1[n] = zero8; }
#pragma unroll
    for (int kt = 0; kt < D_IN / 32; ++kt) {
      v16bf bfr[8];                           // batch all B-frag loads first
#pragma unroll
      for (int n = 0; n < 8; ++n)
        bfr[n] = *(const v16bf*)(w1p +
                     ((size_t)(kt * (D_HID / 16) + g * 8 + n)) * 512 + lane * 16);
      v16bf af0 = load_afrag_f32(arow0 + kt * 32 + khalf);
      v16bf af1 = load_afrag_f32(arow1 + kt * 32 + khalf);
#pragma unroll
      for (int n = 0; n < 8; ++n) {
        a0[n] = __builtin_amdgcn_wmma_f32_16x16x32_bf16(
            false, af0, false, bfr[n], (short)0, a0[n], false, false);
        a1[n] = __builtin_amdgcn_wmma_f32_16x16x32_bf16(
            false, af1, false, bfr[n], (short)0, a1[n], false, false);
      }
    }
    // bias + relu + D-layout -> row-major bf16 tile in per-wave LDS
#pragma unroll
    for (int n = 0; n < 8; ++n) {
      int col = (g * 8 + n) * 16 + l16;
      float bias = b1[col];
#pragma unroll
      for (int r = 0; r < 8; ++r) {
        float v0 = a0[n][r] + bias;
        float v1 = a1[n][r] + bias;
        sc[(r + khalf) * D_HID + col]      = (bf16_t)fmaxf(v0, 0.0f);
        sc[(16 + r + khalf) * D_HID + col] = (bf16_t)fmaxf(v1, 0.0f);
      }
    }
  }
  // same-wave LDS RAW: hardware DScnt ordering (compiler inserts waits)

  // ---------------- layer 2: out[32x128] = h1[32x256] @ w2 + b2 -------------
  v8f c0[D_OUT / 16], c1[D_OUT / 16];
#pragma unroll
  for (int n = 0; n < D_OUT / 16; ++n) { c0[n] = zero8; c1[n] = zero8; }

  const bf16_t* srow0 = sc + l16 * D_HID;
  const bf16_t* srow1 = sc + (16 + l16) * D_HID;
#pragma unroll
  for (int kt = 0; kt < D_HID / 32; ++kt) {
    v16bf bfr[8];                             // batch all B-frag loads first
#pragma unroll
    for (int n = 0; n < 8; ++n)
      bfr[n] = *(const v16bf*)(w2p +
                   ((size_t)(kt * (D_OUT / 16) + n)) * 512 + lane * 16);
    v16bf af0 = load_afrag_lds(srow0 + kt * 32 + khalf);
    v16bf af1 = load_afrag_lds(srow1 + kt * 32 + khalf);
#pragma unroll
    for (int n = 0; n < D_OUT / 16; ++n) {
      c0[n] = __builtin_amdgcn_wmma_f32_16x16x32_bf16(
          false, af0, false, bfr[n], (short)0, c0[n], false, false);
      c1[n] = __builtin_amdgcn_wmma_f32_16x16x32_bf16(
          false, af1, false, bfr[n], (short)0, c1[n], false, false);
    }
  }

#pragma unroll
  for (int n = 0; n < D_OUT / 16; ++n) {
    int col = n * 16 + l16;
    float bias = b2[col];
#pragma unroll
    for (int r = 0; r < 8; ++r) {
      out[(size_t)(row0 + r + khalf) * D_OUT + col] = c0[n][r] + bias;
    }
    if (full) {                               // tail task: skip invalid rows
#pragma unroll
      for (int r = 0; r < 8; ++r) {
        out[(size_t)(row0 + 16 + r + khalf) * D_OUT + col] = c1[n][r] + bias;
      }
    }
  }
}

// ---------------------------------------------------------------------------
extern "C" void kernel_launch(void* const* d_in, const int* in_sizes, int n_in,
                              void* d_out, int out_size, void* d_ws, size_t ws_size,
                              hipStream_t stream) {
  const float* x    = (const float*)d_in[0];
  const int*   esrc = (const int*)d_in[1];
  const int*   edst = (const int*)d_in[2];
  const float* ev   = (const float*)d_in[3];
  const float* eps  = (const float*)d_in[4];
  const float* w1   = (const float*)d_in[5];
  const float* b1   = (const float*)d_in[6];
  const float* w2   = (const float*)d_in[7];
  const float* b2   = (const float*)d_in[8];
  float* out = (float*)d_out;

  // workspace layout: h0 (fp32) | w1 packed bf16 | w2 packed bf16
  char* ws = (char*)d_ws;
  float* h0 = (float*)ws;
  size_t off = (size_t)N_NODES * D_IN * sizeof(float);
  off = (off + 255) & ~(size_t)255;
  bf16_t* w1p = (bf16_t*)(ws + off);
  off += (size_t)(D_IN / 32) * (D_HID / 16) * 512 * sizeof(bf16_t);
  off = (off + 255) & ~(size_t)255;
  bf16_t* w2p = (bf16_t*)(ws + off);

  k_init_h0<<<(N_NODES * D_IN / 4 + 255) / 256, 256, 0, stream>>>(x, eps, h0);
  k_pack_w<<<(D_IN / 32) * (D_HID / 16), 32, 0, stream>>>(w1, D_IN, D_HID, w1p);
  k_pack_w<<<(D_HID / 32) * (D_OUT / 16), 32, 0, stream>>>(w2, D_HID, D_OUT, w2p);
  k_edge_agg<<<AGG_BLOCKS, AGG_THREADS, 0, stream>>>(x, esrc, edst, ev, h0);
  k_mlp<<<(ROW_TASKS + MLP_WPB - 1) / MLP_WPB, 32 * MLP_WPB, 0, stream>>>(
      h0, w1p, b1, w2p, b2, out);
}
